// LinkPredictor_47588237639882
// MI455X (gfx1250) — matrix operand order repaired and checked by reading, not compiled, not run on hardware
//
#include <hip/hip_runtime.h>
#include <hip/hip_bf16.h>
#include <stdint.h>

// LinkPredictor for MI455X (gfx1250, wave32, WMMA).
//
// Strategy:
//   1) Factor the per-edge MLP through per-node transforms:
//        H[n, 0:256]   = W1s @ z[n] + b1      (hs, bias folded in)
//        H[n, 256:512] = W1d @ z[n]           (hd)
//      -> one (100000 x 256) x (256 x 512) GEMM = 26.2 GFLOP (vs 78.6 per-edge).
//   2) Edge phase is pure gather+dot: out[e] = W2 . relu(hs[src]+hd[dst]) + b2
//      614 MB of reads ~ 26 us @ 23.3 TB/s, largely L2-resident (192 MB L2).
//   3) GEMM uses v_wmma_f32_16x16x32_bf16 with hi/lo bf16 split-precision
//      (3 WMMAs per tile: hi*hi + hi*lo + lo*hi, f32 accumulate) for near-f32
//      accuracy on the fast 16-bit matrix pipe.

#define N_NODES 100000
#define N_EDGES 300000
#define IN_CH   256
#define HID     256

typedef __attribute__((ext_vector_type(16))) __bf16 v16bf;
typedef __attribute__((ext_vector_type(8)))  __bf16 v8bf;
typedef __attribute__((ext_vector_type(8)))  float  v8f;

static __device__ __forceinline__ unsigned short f32_to_bf16_rne(float x) {
    unsigned u = __builtin_bit_cast(unsigned, x);
    u += 0x7FFFu + ((u >> 16) & 1u);          // round-to-nearest-even
    return (unsigned short)(u >> 16);
}
static __device__ __forceinline__ float bf16_to_f32(unsigned short h) {
    return __builtin_bit_cast(float, ((unsigned)h) << 16);
}

// ---------------------------------------------------------------------------
// Pack z (f32, row-major 100000x256) into bf16 hi/lo arrays (row-major).
// ---------------------------------------------------------------------------
__global__ __launch_bounds__(256) void pack_z_kernel(const float* __restrict__ z,
                                                     __bf16* __restrict__ zh,
                                                     __bf16* __restrict__ zl) {
    size_t i = (size_t)blockIdx.x * 256 + threadIdx.x;   // < N_NODES*IN_CH
    float v = z[i];
    unsigned short hi = f32_to_bf16_rne(v);
    unsigned short lo = f32_to_bf16_rne(v - bf16_to_f32(hi));
    zh[i] = __builtin_bit_cast(__bf16, hi);
    zl[i] = __builtin_bit_cast(__bf16, lo);
}

// ---------------------------------------------------------------------------
// Pack W1 (f32, HIDx512 row-major) into per-lane B-fragment order so the GEMM
// loads each 16x16-col B fragment slice as one contiguous 32B v16bf.
// Layout: wpk[((half*16 + nTile)*8 + kStep)*32 + lane][16]
//   B fragment (32x16, 16-bit): col N = lane&15, K = ((lane<16)?0:8)+j+((j>=8)?8:0), j=0..15
//   Source: B[k][n] = W1[hid*512 + half*256 + k],  hid = nTile*16 + (lane&15)
// ---------------------------------------------------------------------------
__global__ __launch_bounds__(256) void pack_w_kernel(const float* __restrict__ W1,
                                                     __bf16* __restrict__ wh,
                                                     __bf16* __restrict__ wl) {
    const int id   = blockIdx.x * 256 + threadIdx.x;     // < 131072
    const int j    = id & 15;
    const int lane = (id >> 4) & 31;
    const int ks   = (id >> 9) & 7;
    const int t    = (id >> 12) & 15;
    const int half = (id >> 16) & 1;
    const int c    = lane & 15;
    const int kOff = (lane >> 4) << 3;
    const int hid  = t * 16 + c;
    const int k    = ks * 32 + kOff + j + (j & 8);
    const float v  = W1[(size_t)hid * 512 + half * 256 + k];
    unsigned short hi = f32_to_bf16_rne(v);
    unsigned short lo = f32_to_bf16_rne(v - bf16_to_f32(hi));
    wh[id] = __builtin_bit_cast(__bf16, hi);
    wl[id] = __builtin_bit_cast(__bf16, lo);
}

// ---------------------------------------------------------------------------
// Node GEMM: H = z @ [W1s^T | W1d^T]  (+b1 folded into the hs half).
// grid = (N_NODES/16, 2), block = 128 (4 waves). Each wave owns a 16x64 tile:
// 4 accumulators, K looped in 8 steps of 32 with 3 split-precision WMMAs each.
// A fragment (16x32 bf16): row = lane&15; lanes<16 hold K {0-7,16-23} (+kb=0),
// lanes>=16 hold K {8-15,24-31} (+kb=8) -> two contiguous 16B loads per part.
// ---------------------------------------------------------------------------
__global__ __launch_bounds__(128) void node_gemm_kernel(
    const __bf16* __restrict__ zh, const __bf16* __restrict__ zl,
    const __bf16* __restrict__ wh, const __bf16* __restrict__ wl,
    const float* __restrict__ b1, float* __restrict__ H)
{
    const int lane = threadIdx.x & 31;
    const int wv   = threadIdx.x >> 5;        // 0..3
    const int m0   = blockIdx.x << 4;
    const int half = blockIdx.y;              // 0 -> hs(+b1), 1 -> hd

    const int arow = m0 + (lane & 15);
    const int kb   = (lane >> 4) << 3;        // 0 or 8

    const v8f zero = {0.f, 0.f, 0.f, 0.f, 0.f, 0.f, 0.f, 0.f};
    v8f acc[4];
    #pragma unroll
    for (int t = 0; t < 4; ++t) acc[t] = zero;

    const __bf16* zhp = zh + (size_t)arow * IN_CH + kb;
    const __bf16* zlp = zl + (size_t)arow * IN_CH + kb;

    #pragma unroll
    for (int ks = 0; ks < 8; ++ks) {
        const int k0 = ks * 32;
        v8bf p, q;
        p = *(const v8bf*)(zhp + k0);
        q = *(const v8bf*)(zhp + k0 + 16);
        v16bf ah = __builtin_shufflevector(p, q, 0,1,2,3,4,5,6,7,8,9,10,11,12,13,14,15);
        p = *(const v8bf*)(zlp + k0);
        q = *(const v8bf*)(zlp + k0 + 16);
        v16bf al = __builtin_shufflevector(p, q, 0,1,2,3,4,5,6,7,8,9,10,11,12,13,14,15);

        #pragma unroll
        for (int t2 = 0; t2 < 4; ++t2) {
            const int t = (wv << 2) + t2;
            const size_t bidx = ((((size_t)half * 16 + t) * 8 + ks) * 32 + lane) * 16;
            v16bf bh = *(const v16bf*)(wh + bidx);
            v16bf bl = *(const v16bf*)(wl + bidx);
            // split-precision: a*b ~= ah*bh + ah*bl + al*bh (f32 accumulate)
            acc[t2] = __builtin_amdgcn_wmma_f32_16x16x32_bf16(
                          false, ah, false, bh, (short)0, acc[t2], false, false);
            acc[t2] = __builtin_amdgcn_wmma_f32_16x16x32_bf16(
                          false, ah, false, bl, (short)0, acc[t2], false, false);
            acc[t2] = __builtin_amdgcn_wmma_f32_16x16x32_bf16(
                          false, al, false, bh, (short)0, acc[t2], false, false);
        }
    }

    // C/D layout: col N = lane&15; element r -> row M = r + ((lane>=16)?8:0)
    const int cN = lane & 15;
    const int rb = (lane >> 4) << 3;
    #pragma unroll
    for (int t2 = 0; t2 < 4; ++t2) {
        const int n    = (half << 8) + ((((wv << 2) + t2) << 4) + cN);
        const float bias = half ? 0.0f : b1[n];   // uniform branch, EXEC stays full
        float* dst = H + (size_t)(m0 + rb) * (2 * HID) + n;
        #pragma unroll
        for (int r = 0; r < 8; ++r)
            dst[(size_t)r * (2 * HID)] = acc[t2][r] + bias;
    }
}

// ---------------------------------------------------------------------------
// Edge phase: one wave32 per edge. lane handles 8 channels (2x float4 each
// from hs[src] and hd[dst]), ReLU + dot with W2, __shfl_xor wave reduction.
// ---------------------------------------------------------------------------
__global__ __launch_bounds__(256) void edge_mlp_kernel(
    const float* __restrict__ H, const int* __restrict__ ei,
    const float* __restrict__ W2, const float* __restrict__ b2,
    float* __restrict__ out)
{
    const int lane = threadIdx.x & 31;
    const int e    = blockIdx.x * 8 + (threadIdx.x >> 5);
    const int s = ei[e];
    const int d = ei[N_EDGES + e];
    const float4* hs = (const float4*)(H + (size_t)s * 512) + lane * 2;
    const float4* hd = (const float4*)(H + (size_t)d * 512 + 256) + lane * 2;
    const float4* w  = (const float4*)W2 + lane * 2;
    float4 s0 = hs[0], s1 = hs[1];
    float4 d0 = hd[0], d1 = hd[1];
    float4 w0 = w[0],  w1 = w[1];
    float acc =
        w0.x * fmaxf(s0.x + d0.x, 0.f) + w0.y * fmaxf(s0.y + d0.y, 0.f) +
        w0.z * fmaxf(s0.z + d0.z, 0.f) + w0.w * fmaxf(s0.w + d0.w, 0.f) +
        w1.x * fmaxf(s1.x + d1.x, 0.f) + w1.y * fmaxf(s1.y + d1.y, 0.f) +
        w1.z * fmaxf(s1.z + d1.z, 0.f) + w1.w * fmaxf(s1.w + d1.w, 0.f);
    #pragma unroll
    for (int off = 16; off > 0; off >>= 1)
        acc += __shfl_xor(acc, off, 32);
    if (lane == 0) out[e] = acc + b2[0];
}

// ---------------------------------------------------------------------------
// Launch. Inputs (reference order): z, W1, b1, W2, b2, edge_index.
// Workspace layout (all offsets 32B-aligned):
//   H  : f32  [100000 x 512]                204,800,000 B
//   zh : bf16 [100000 x 256]                 51,200,000 B
//   zl : bf16 [100000 x 256]                 51,200,000 B
//   wh : bf16 [131072]  (B-fragment order)      262,144 B
//   wl : bf16 [131072]                          262,144 B
// ---------------------------------------------------------------------------
extern "C" void kernel_launch(void* const* d_in, const int* in_sizes, int n_in,
                              void* d_out, int out_size, void* d_ws, size_t ws_size,
                              hipStream_t stream)
{
    const float* z  = (const float*)d_in[0];
    const float* W1 = (const float*)d_in[1];
    const float* b1 = (const float*)d_in[2];
    const float* W2 = (const float*)d_in[3];
    const float* b2 = (const float*)d_in[4];
    const int*   ei = (const int*)d_in[5];   // edge_index as int32 per harness

    char*   ws = (char*)d_ws;
    float*  H  = (float*)(ws);
    __bf16* zh = (__bf16*)(ws + 204800000);
    __bf16* zl = (__bf16*)(ws + 256000000);
    __bf16* wh = (__bf16*)(ws + 307200000);
    __bf16* wl = (__bf16*)(ws + 307462144);

    pack_z_kernel<<<(N_NODES * IN_CH) / 256, 256, 0, stream>>>(z, zh, zl);
    pack_w_kernel<<<131072 / 256, 256, 0, stream>>>(W1, wh, wl);

    dim3 gGemm(N_NODES / 16, 2);
    node_gemm_kernel<<<gGemm, 128, 0, stream>>>(zh, zl, wh, wl, b1, H);

    edge_mlp_kernel<<<N_EDGES / 8, 256, 0, stream>>>(H, ei, W2, b2, (float*)d_out);
}
